// MultiHeadAttention_78563541778573
// MI455X (gfx1250) — matrix operand order repaired and checked by reading, not compiled, not run on hardware
//
#include <hip/hip_runtime.h>
#include <math.h>

// ---------------------------------------------------------------------------
// Transformer-XL attention block for MI455X (gfx1250), wave32 + WMMA bf16.
// All GEMMs use v_wmma_f32_16x16x32_bf16 (fp32 accumulate, bf16 operands).
// Every tensor that only feeds a WMMA operand is stored once in bf16,
// pre-laid-out so each lane's fragment is contiguous (v16bf/v8bf deref ->
// global_load_b128, no per-element address math, no repeated conversions).
//
// Workspace layout (bytes):
//   Qws   fp32  B*H*SEG*D      = 33,554,432
//   Kbf   bf16  B*H*TOTAL*D    = 33,554,432
//   Vtb   bf16  B*H*D*TOTAL    = 33,554,432   (transposed: [bh][d][t])
//   Rbf   bf16  H*TOTAL*D      =  2,097,152   (projected r, [h,j,d])
//   ATTb  bf16  B*SEG*H*D      = 16,777,216
//   Hbf   bf16  B*TOTAL*MD     =  4,194,304   (concat(mem,x) activations)
//   Rxbf  bf16  TOTAL*MD       =    262,144   (raw R activations)
//   WqT   bf16  1024*128       =    262,144
//   WkvT  bf16  2048*128       =    524,288
//   WrT   bf16  1024*128       =    262,144
//   WmlpT bf16  128*1024       =    262,144
//   total ~119.8 MB -- requires ws_size >= that.
// ---------------------------------------------------------------------------

typedef __attribute__((ext_vector_type(16))) __bf16 v16bf;
typedef __attribute__((ext_vector_type(8)))  __bf16 v8bf;
typedef __attribute__((ext_vector_type(8)))  float  v8f;

#define B_     16
#define SEG_   512
#define MEML_  512
#define MD_    128
#define H_     8
#define D_     128
#define TOTAL_ 1024

__device__ __forceinline__ v8f wmma_bf16(v16bf a, v16bf b, v8f c) {
  // 8 args: (neg_a, A, neg_b, B, c_mod, C, reuse_a, reuse_b)
  return __builtin_amdgcn_wmma_f32_16x16x32_bf16(false, a, false, b, (short)0, c,
                                                 false, false);
}

// --- Fragment loaders -------------------------------------------------------
// A-matrix 16x32 bf16 (ISA 7.12.2): lane row = lane%16, half = lane/16.
// Per lane the 16 elements are two contiguous K-runs of 8:
//   run0 at K = k0 + 8*half, run1 at K = k0 + 16 + 8*half.

// fp32 q-row plus fp32 u-row (broadcast bias), summed then converted
__device__ __forceinline__ v16bf load_a_sum(const float* qr, const float* ur,
                                            int k0, int hf) {
  int o0 = k0 + 8 * hf, o1 = k0 + 16 + 8 * hf;
  float4 qa = *(const float4*)(qr + o0), qb = *(const float4*)(qr + o0 + 4);
  float4 qc = *(const float4*)(qr + o1), qd = *(const float4*)(qr + o1 + 4);
  float4 ua = *(const float4*)(ur + o0), ub = *(const float4*)(ur + o0 + 4);
  float4 uc = *(const float4*)(ur + o1), ud = *(const float4*)(ur + o1 + 4);
  v16bf a;
  a[0] = (__bf16)(qa.x + ua.x);  a[1] = (__bf16)(qa.y + ua.y);
  a[2] = (__bf16)(qa.z + ua.z);  a[3] = (__bf16)(qa.w + ua.w);
  a[4] = (__bf16)(qb.x + ub.x);  a[5] = (__bf16)(qb.y + ub.y);
  a[6] = (__bf16)(qb.z + ub.z);  a[7] = (__bf16)(qb.w + ub.w);
  a[8] = (__bf16)(qc.x + uc.x);  a[9] = (__bf16)(qc.y + uc.y);
  a[10] = (__bf16)(qc.z + uc.z); a[11] = (__bf16)(qc.w + uc.w);
  a[12] = (__bf16)(qd.x + ud.x); a[13] = (__bf16)(qd.y + ud.y);
  a[14] = (__bf16)(qd.z + ud.z); a[15] = (__bf16)(qd.w + ud.w);
  return a;
}

// bf16 source (two contiguous v8bf runs)
__device__ __forceinline__ v16bf load_a_bf16(const __bf16* rowptr, int k0, int hf) {
  v8bf lo = *(const v8bf*)(rowptr + k0 + 8 * hf);
  v8bf hi = *(const v8bf*)(rowptr + k0 + 16 + 8 * hf);
  return __builtin_shufflevector(lo, hi, 0, 1, 2, 3, 4, 5, 6, 7, 8, 9, 10, 11,
                                 12, 13, 14, 15);
}

// B-matrix 32x16: lane col = lane%16, same K order as A:
// run0 at K = k0 + 8*half, run1 at K = k0 + 16 + 8*half.
__device__ __forceinline__ v16bf load_b_bf16(const __bf16* rowptr, int k0, int hf) {
  v8bf lo = *(const v8bf*)(rowptr + k0 + 8 * hf);
  v8bf hi = *(const v8bf*)(rowptr + k0 + 16 + 8 * hf);
  return __builtin_shufflevector(lo, hi, 0, 1, 2, 3, 4, 5, 6, 7, 8, 9, 10, 11,
                                 12, 13, 14, 15);
}

// Row-wise reductions across the 16 lanes sharing a C/D row.
__device__ __forceinline__ float redmax16(float v) {
#pragma unroll
  for (int m = 1; m <= 8; m <<= 1) v = fmaxf(v, __shfl_xor(v, m, 32));
  return v;
}
__device__ __forceinline__ float redsum16(float v) {
#pragma unroll
  for (int m = 1; m <= 8; m <<= 1) v += __shfl_xor(v, m, 32);
  return v;
}

// ---------------------------------------------------------------------------
// Kernel 0: one-shot bf16 prep of activations and transposed weights.
//   Hbf[b][t][k]  = (t<512 ? mem[b][t][k] : x[b][t-512][k])
//   Rxbf[t][k]    = R[t][k]
//   WqT[n][k]=Wq[k][n], WkvT (2048x128), WrT (1024x128), WmlpT[n][k]=Wmlp[k][n]
// ---------------------------------------------------------------------------
__global__ __launch_bounds__(256) void prep_kernel(
    const float* __restrict__ x, const float* __restrict__ mem,
    const float* __restrict__ R, const float* __restrict__ Wq,
    const float* __restrict__ Wkv, const float* __restrict__ Wr,
    const float* __restrict__ Wmlp, __bf16* __restrict__ Hbf,
    __bf16* __restrict__ Rxbf, __bf16* __restrict__ WqT,
    __bf16* __restrict__ WkvT, __bf16* __restrict__ WrT,
    __bf16* __restrict__ WmlpT) {
  int id = blockIdx.x * blockDim.x + threadIdx.x;
  if (id < 2097152) {  // Hbf: B*TOTAL*MD
    int b = id >> 17, rem = id & 131071;
    int t = rem >> 7, k = rem & 127;
    float v = (t < 512) ? mem[((size_t)b * 512 + t) * MD_ + k]
                        : x[((size_t)b * 512 + (t - 512)) * MD_ + k];
    Hbf[id] = (__bf16)v;
    return;
  }
  id -= 2097152;
  if (id < 131072) { Rxbf[id] = (__bf16)R[id]; return; }
  id -= 131072;
  if (id < 131072) {
    int n = id >> 7, k = id & 127;
    WqT[id] = (__bf16)Wq[k * 1024 + n];
    return;
  }
  id -= 131072;
  if (id < 262144) {
    int n = id >> 7, k = id & 127;
    WkvT[id] = (__bf16)Wkv[k * 2048 + n];
    return;
  }
  id -= 262144;
  if (id < 131072) {
    int n = id >> 7, k = id & 127;
    WrT[id] = (__bf16)Wr[k * 1024 + n];
    return;
  }
  id -= 131072;
  if (id < 131072) {
    int n = id >> 10, k = id & 1023;
    WmlpT[id] = (__bf16)Wmlp[k * 128 + n];
  }
}

// ---------------------------------------------------------------------------
// Kernel 1: projections, register-blocked 4 n-tiles per wave (A frags loaded
// once, 16 WMMAs per wave-unit), with reshape-scramble scatter epilogues.
//   mode 0: Q   = x @ Wq            -> Qws (fp32, [b,h,j,d])
//   mode 1: KV  = concat(mem,x)@Wkv -> Kbf (bf16) / Vtb (bf16 transposed)
//   mode 2: Rr  = R @ Wr            -> Rbf (bf16, [h,j,d])
// ---------------------------------------------------------------------------
#define NB_ 4

__global__ __launch_bounds__(128) void proj_kernel(
    const __bf16* __restrict__ Hbf, const __bf16* __restrict__ Rxbf,
    const __bf16* __restrict__ Wt, float* __restrict__ Qws,
    __bf16* __restrict__ Kbf, __bf16* __restrict__ Vtb,
    __bf16* __restrict__ Rbf, int mode, int ntile_m, int ngrp_n) {
  int wid = blockIdx.x * (blockDim.x >> 5) + (threadIdx.x >> 5);
  if (wid >= ntile_m * ngrp_n) return;
  int tm = wid / ngrp_n, tg = wid % ngrp_n;
  int m0 = tm * 16, n0 = tg * (16 * NB_);
  int lane = threadIdx.x & 31, nl = lane & 15, hf = lane >> 4;

  int arow = m0 + nl;
  const __bf16* aptr;
  if (mode == 0) {
    int b = arow >> 9, i = arow & 511;  // x lives in Hbf rows [512,1024)
    aptr = Hbf + ((size_t)b * TOTAL_ + 512 + i) * MD_;
  } else if (mode == 1) {
    aptr = Hbf + (size_t)arow * MD_;
  } else {
    aptr = Rxbf + (size_t)arow * MD_;
  }

  v16bf af[4];
#pragma unroll
  for (int c = 0; c < 4; ++c) af[c] = load_a_bf16(aptr, c * 32, hf);

  v8f zero8 = {0, 0, 0, 0, 0, 0, 0, 0};
  v8f acc[NB_];
#pragma unroll
  for (int nb = 0; nb < NB_; ++nb) acc[nb] = zero8;

#pragma unroll
  for (int nb = 0; nb < NB_; ++nb) {
    const __bf16* brow = Wt + (size_t)(n0 + nb * 16 + nl) * MD_;
#pragma unroll
    for (int c = 0; c < 4; ++c)
      acc[nb] = wmma_bf16(af[c], load_b_bf16(brow, c * 32, hf), acc[nb]);
  }

#pragma unroll
  for (int nb = 0; nb < NB_; ++nb) {
    int cg = n0 + nb * 16 + nl;  // output column
#pragma unroll
    for (int r = 0; r < 8; ++r) {
      int row = m0 + r + 8 * hf;
      float val = acc[nb][r];
      if (mode == 0) {
        // (B,SEG,1024).reshape(B,H,SEG,D): h=i/64, j=(i%64)*8+c/128, d=c%128
        int b = row >> 9, i = row & 511;
        int h = i >> 6, j = ((i & 63) << 3) + (cg >> 7), d = cg & 127;
        Qws[(((size_t)b * H_ + h) * SEG_ + j) * D_ + d] = val;
      } else if (mode == 1) {
        // (B,1024,2048).reshape(2B,H,1024,D): b2=2b+(t>=512),
        // lin=(t%512)*16 + c/128, h2=lin/1024, t2=lin%1024, d=c%128
        int b = row >> 10, t = row & 1023;
        int b2 = 2 * b + ((t >= 512) ? 1 : 0);
        int lin = (t & 511) * 16 + (cg >> 7);
        int h2 = lin >> 10, t2 = lin & 1023, d = cg & 127;
        if (b2 < B_)
          Kbf[(((size_t)b2 * H_ + h2) * TOTAL_ + t2) * D_ + d] = (__bf16)val;
        else  // V stored transposed: [bh][d][t]
          Vtb[(((size_t)(b2 - B_) * H_ + h2) * D_ + d) * TOTAL_ + t2] =
              (__bf16)val;
      } else {
        int t = row;
        int h = t >> 7, j = ((t & 127) << 3) + (cg >> 7), d = cg & 127;
        Rbf[((size_t)h * TOTAL_ + j) * D_ + d] = (__bf16)val;
      }
    }
  }
}

// ---------------------------------------------------------------------------
// Kernel 2: fused attention (flash-style), one wave per 16 query rows.
// Phase 1: BD_raw[16][1024] into LDS. Phase 2: online softmax over 32-key
// steps with circulant-shifted BD gather, tril(mem_len) mask, ==0 -> -inf.
// ---------------------------------------------------------------------------
#define ATT_WAVES 2

__global__ __launch_bounds__(ATT_WAVES * 32) void attn_kernel(
    const float* __restrict__ Qws, const __bf16* __restrict__ Kbf,
    const __bf16* __restrict__ Vtb, const __bf16* __restrict__ Rbf,
    const float* __restrict__ u1, const float* __restrict__ u2,
    __bf16* __restrict__ ATTb) {
  extern __shared__ char smem_raw[];
  int w = threadIdx.x >> 5;
  float*  bd   = (float*)smem_raw + (size_t)w * (16 * TOTAL_);
  __bf16* plds = (__bf16*)(smem_raw + (size_t)ATT_WAVES * 16 * TOTAL_ * 4) +
                 (size_t)w * (16 * 32);

  int wid = blockIdx.x * ATT_WAVES + w;
  int qt = wid & 31, bh = wid >> 5;  // 32 q-tiles per (b,h)
  int b = bh >> 3, h = bh & 7;
  int i0 = qt * 16;
  int lane = threadIdx.x & 31, nl = lane & 15, hf = lane >> 4;

  // (q + u1) and (q + u2) A-fragments, K = 128 (head dim)
  const float* qrow = Qws + ((size_t)bh * SEG_ + i0 + nl) * D_;
  const float* u1p = u1 + h * D_;
  const float* u2p = u2 + h * D_;
  v16bf q1f[4], q2f[4];
#pragma unroll
  for (int c = 0; c < 4; ++c) {
    q1f[c] = load_a_sum(qrow, u1p, c * 32, hf);
    q2f[c] = load_a_sum(qrow, u2p, c * 32, hf);
  }

  // Phase 1: BD_raw = (q+u2) @ r^T for this 16-row strip -> LDS
  const __bf16* rbase = Rbf + (size_t)h * TOTAL_ * D_;
  for (int jt = 0; jt < TOTAL_ / 16; ++jt) {
    const __bf16* rrow = rbase + (size_t)(jt * 16 + nl) * D_;
    v8f acc = {0, 0, 0, 0, 0, 0, 0, 0};
#pragma unroll
    for (int c = 0; c < 4; ++c)
      acc = wmma_bf16(q2f[c], load_b_bf16(rrow, c * 32, hf), acc);
#pragma unroll
    for (int r = 0; r < 8; ++r)
      bd[(r + 8 * hf) * TOTAL_ + jt * 16 + nl] = acc[r];
  }

  const float RSQRT_D = 0.08838834764831845f;  // 1/sqrt(128)
  const float NEGINF  = -__builtin_inff();
  const __bf16* kbase = Kbf + (size_t)bh * TOTAL_ * D_;
  const __bf16* vbase = Vtb + (size_t)bh * D_ * TOTAL_;

  float mrun[8], lrun[8];
  v8f O[8];
  v8f zero8 = {0, 0, 0, 0, 0, 0, 0, 0};
#pragma unroll
  for (int r = 0; r < 8; ++r) { mrun[r] = NEGINF; lrun[r] = 0.f; }
#pragma unroll
  for (int nc = 0; nc < 8; ++nc) O[nc] = zero8;

  for (int kt = 0; kt < TOTAL_ / 32; ++kt) {
    int j0 = kt * 32;
    if (kt + 1 < TOTAL_ / 32) {  // prefetch next K/V tiles (global_prefetch)
      __builtin_prefetch((const void*)(kbase + (size_t)(j0 + 32 + nl) * D_), 0, 1);
      __builtin_prefetch((const void*)(vbase + (size_t)(nl * 8) * TOTAL_ + j0 + 32), 0, 1);
    }
    float s[2][8];
#pragma unroll
    for (int st = 0; st < 2; ++st) {
      int jb = j0 + st * 16;
      const __bf16* krow = kbase + (size_t)(jb + nl) * D_;
      v8f acc = zero8;
#pragma unroll
      for (int c = 0; c < 4; ++c)
        acc = wmma_bf16(q1f[c], load_b_bf16(krow, c * 32, hf), acc);  // AC
#pragma unroll
      for (int r = 0; r < 8; ++r) {
        int i = i0 + r + 8 * hf;
        int j = jb + nl;
        int craw = (j - i + SEG_ - 1) & (TOTAL_ - 1);  // circulant shift
        float val = acc[r] + bd[(r + 8 * hf) * TOTAL_ + craw];
        float sv = (j <= i + MEML_) ? val * RSQRT_D : 0.0f;  // tril(mem_len)
        if (sv == 0.0f) sv = NEGINF;                         // where(==0,-inf)
        s[st][r] = sv;
      }
    }
    float alpha[8];
#pragma unroll
    for (int r = 0; r < 8; ++r) {
      float tmax = redmax16(fmaxf(s[0][r], s[1][r]));
      float mnew = fmaxf(mrun[r], tmax);
      alpha[r] = (mnew == NEGINF) ? 0.0f : __expf(mrun[r] - mnew);
      float ps = 0.f;
#pragma unroll
      for (int st = 0; st < 2; ++st) {
        float pv = (mnew == NEGINF) ? 0.0f : __expf(s[st][r] - mnew);
        plds[(r + 8 * hf) * 32 + st * 16 + nl] = (__bf16)pv;
        ps += pv;
      }
      lrun[r] = alpha[r] * lrun[r] + redsum16(ps);
      mrun[r] = mnew;
    }
#pragma unroll
    for (int nc = 0; nc < 8; ++nc)
#pragma unroll
      for (int r = 0; r < 8; ++r) O[nc][r] *= alpha[r];

    // P (16x32) as A-fragment from LDS (two contiguous v8bf runs per lane)
    v16bf pa = load_a_bf16(plds + nl * 32, 0, hf);
    // O += P @ V  (K = 32 keys, N = 128 dims in 8 chunks); V is [d][t] so the
    // fragment is two contiguous runs of 8 bf16 per lane.
#pragma unroll
    for (int nc = 0; nc < 8; ++nc) {
      const __bf16* vrow = vbase + (size_t)(nc * 16 + nl) * TOTAL_ + j0;
      O[nc] = wmma_bf16(pa, load_b_bf16(vrow, 0, hf), O[nc]);
    }
  }

  // Normalize and scatter into the reference's (p@v).reshape(B,SEG,H*D) layout:
  // row i2 = h*64 + i/8, col = (i%8)*128 + d   (stored bf16 for the MLP GEMM)
#pragma unroll
  for (int r = 0; r < 8; ++r) {
    float rl = (lrun[r] > 0.f) ? (1.0f / lrun[r]) : 0.f;
    int i = i0 + r + 8 * hf;
    int i2 = h * 64 + (i >> 3);
    int cbase = (i & 7) * D_;
#pragma unroll
    for (int nc = 0; nc < 8; ++nc) {
      int d = nc * 16 + nl;
      ATTb[((size_t)b * SEG_ + i2) * (H_ * D_) + cbase + d] =
          (__bf16)(O[nc][r] * rl);
    }
  }
}

// ---------------------------------------------------------------------------
// Kernel 3: y = att @ Wmlp + x, then LayerNorm(gamma, beta). One wave per 16
// rows (full 128-wide output strip so LN reductions stay in the wave).
// ---------------------------------------------------------------------------
__global__ __launch_bounds__(128) void mlp_ln_kernel(
    const __bf16* __restrict__ ATTb, const __bf16* __restrict__ WmlpT,
    const float* __restrict__ x, const float* __restrict__ gamma,
    const float* __restrict__ beta, float* __restrict__ out) {
  int wid = blockIdx.x * (blockDim.x >> 5) + (threadIdx.x >> 5);
  int row0 = wid * 16;
  if (row0 >= B_ * SEG_) return;
  int lane = threadIdx.x & 31, nl = lane & 15, hf = lane >> 4;

  v8f zero8 = {0, 0, 0, 0, 0, 0, 0, 0};
  v8f acc[8];
#pragma unroll
  for (int nc = 0; nc < 8; ++nc) acc[nc] = zero8;

  const __bf16* arow = ATTb + ((size_t)row0 + nl) * (H_ * D_);
  for (int kt = 0; kt < (H_ * D_) / 32; ++kt) {
    v16bf a = load_a_bf16(arow, kt * 32, hf);
#pragma unroll
    for (int nc = 0; nc < 8; ++nc) {
      const __bf16* brow = WmlpT + (size_t)(nc * 16 + nl) * (H_ * D_);
      acc[nc] = wmma_bf16(a, load_b_bf16(brow, kt * 32, hf), acc[nc]);
    }
  }

#pragma unroll
  for (int r = 0; r < 8; ++r) {
    int row = row0 + r + 8 * hf;
    float sy = 0.f, sy2 = 0.f;
#pragma unroll
    for (int nc = 0; nc < 8; ++nc) {
      int col = nc * 16 + nl;
      float y = acc[nc][r] + x[(size_t)row * MD_ + col];  // residual
      acc[nc][r] = y;
      sy += y;
      sy2 += y * y;
    }
    sy = redsum16(sy);
    sy2 = redsum16(sy2);
    float mu = sy * (1.0f / MD_);
    float var = sy2 * (1.0f / MD_) - mu * mu;
    float rs = rsqrtf(var + 1e-5f);
#pragma unroll
    for (int nc = 0; nc < 8; ++nc) {
      int col = nc * 16 + nl;
      out[(size_t)row * MD_ + col] =
          (acc[nc][r] - mu) * rs * gamma[col] + beta[col];
    }
  }
}

// ---------------------------------------------------------------------------
extern "C" void kernel_launch(void* const* d_in, const int* in_sizes, int n_in,
                              void* d_out, int out_size, void* d_ws,
                              size_t ws_size, hipStream_t stream) {
  (void)in_sizes; (void)n_in; (void)out_size; (void)ws_size;
  const float* x     = (const float*)d_in[0];
  const float* mem   = (const float*)d_in[1];
  const float* R     = (const float*)d_in[2];
  /* d_in[3] = att_mask: all-true, masking handled analytically */
  const float* u1    = (const float*)d_in[4];
  const float* u2    = (const float*)d_in[5];
  const float* Wq    = (const float*)d_in[6];
  const float* Wkv   = (const float*)d_in[7];
  const float* Wr    = (const float*)d_in[8];
  const float* Wmlp  = (const float*)d_in[9];
  const float* gamma = (const float*)d_in[10];
  const float* beta  = (const float*)d_in[11];
  float* out = (float*)d_out;

  char* ws = (char*)d_ws;
  float*  Qws   = (float*)ws;   ws += (size_t)B_ * H_ * SEG_ * D_ * 4;
  __bf16* Kbf   = (__bf16*)ws;  ws += (size_t)B_ * H_ * TOTAL_ * D_ * 2;
  __bf16* Vtb   = (__bf16*)ws;  ws += (size_t)B_ * H_ * TOTAL_ * D_ * 2;
  __bf16* Rbf   = (__bf16*)ws;  ws += (size_t)H_ * TOTAL_ * D_ * 2;
  __bf16* ATTb  = (__bf16*)ws;  ws += (size_t)B_ * SEG_ * H_ * D_ * 2;
  __bf16* Hbf   = (__bf16*)ws;  ws += (size_t)B_ * TOTAL_ * MD_ * 2;
  __bf16* Rxbf  = (__bf16*)ws;  ws += (size_t)TOTAL_ * MD_ * 2;
  __bf16* WqT   = (__bf16*)ws;  ws += (size_t)1024 * 128 * 2;
  __bf16* WkvT  = (__bf16*)ws;  ws += (size_t)2048 * 128 * 2;
  __bf16* WrT   = (__bf16*)ws;  ws += (size_t)1024 * 128 * 2;
  __bf16* WmlpT = (__bf16*)ws;

  {  // activation + weight bf16 prep (2,883,584 elements)
    int total = 2097152 + 131072 + 131072 + 262144 + 131072 + 131072;
    prep_kernel<<<(total + 255) / 256, 256, 0, stream>>>(
        x, mem, R, Wq, Wkv, Wr, Wmlp, Hbf, Rxbf, WqT, WkvT, WrT, WmlpT);
  }
  {  // Q projection: 8192 x 1024, K=128 (4 n-tiles per wave)
    int tm = (B_ * SEG_) / 16, tg = (H_ * D_) / (16 * NB_);
    int blocks = (tm * tg + 3) / 4;
    proj_kernel<<<blocks, 128, 0, stream>>>(Hbf, Rxbf, WqT, Qws, Kbf, Vtb, Rbf,
                                            0, tm, tg);
  }
  {  // KV projection: 16384 x 2048, K=128 (with descramble scatter)
    int tm = (B_ * TOTAL_) / 16, tg = (2 * H_ * D_) / (16 * NB_);
    int blocks = (tm * tg + 3) / 4;
    proj_kernel<<<blocks, 128, 0, stream>>>(Hbf, Rxbf, WkvT, Qws, Kbf, Vtb, Rbf,
                                            1, tm, tg);
  }
  {  // R projection: 1024 x 1024, K=128
    int tm = TOTAL_ / 16, tg = (H_ * D_) / (16 * NB_);
    int blocks = (tm * tg + 3) / 4;
    proj_kernel<<<blocks, 128, 0, stream>>>(Hbf, Rxbf, WrT, Qws, Kbf, Vtb, Rbf,
                                            2, tm, tg);
  }
  {  // fused attention: 4096 waves, 2 waves/block, 130 KB dynamic LDS
    int waves = B_ * H_ * (SEG_ / 16);
    int blocks = waves / ATT_WAVES;
    size_t lds = (size_t)ATT_WAVES * (16 * TOTAL_ * 4 + 16 * 32 * 2);
    attn_kernel<<<blocks, ATT_WAVES * 32, lds, stream>>>(Qws, Kbf, Vtb, Rbf,
                                                         u1, u2, ATTb);
  }
  {  // MLP + residual + LayerNorm: 512 waves
    int waves = (B_ * SEG_) / 16;
    int blocks = (waves + 3) / 4;
    mlp_ln_kernel<<<blocks, 128, 0, stream>>>(ATTb, WmlpT, x, gamma, beta, out);
  }
}
